// SpectralPolarityMixer_18365280157943
// MI455X (gfx1250) — matrix operand order repaired and checked by reading, not compiled, not run on hardware
//
#include <hip/hip_runtime.h>

// SpectralPolarityMixer on gfx1250 (MI455X).
// B=H=M=16, N=2048, DH=64.  Truncated-rfft2 -> per-(b,h) filter -> irfft2 is a
// linear operator: out = A3 (2048x32) * [ mode-space 16x16 filter ] * A1 (32x2048) * q.
// Heavy stages are bf16 WMMA (3-term hi/lo split, f32 accumulate); middle is tiny VALU.
// Memory-bound: ~512MB traffic -> ~22us floor @ 23.3 TB/s; compute ~26 GFLOP << roof.

typedef __attribute__((ext_vector_type(16))) __bf16 v16bf;
typedef __attribute__((ext_vector_type(8)))  __bf16 bf16x8;
typedef __attribute__((ext_vector_type(8)))  float  v8f;

#define SEQ_N   2048
#define DH_N    64
#define NMODE   16
#define TWO_PI  6.283185307179586f

// ---- WMMA fragment loaders -------------------------------------------------
// A operand (16x32 bf16), contiguous k order chosen here: lane row = lane&15; lanes 0-15 hold
// K = 0..15, lanes 16-31 hold K = 16..31.  With p pointing at
// rowBase + (lane>=16 ? 16 : 0), the two 8-element chunks sit at p and p+8.
__device__ __forceinline__ v16bf ld_frag_a(const __bf16* p) {
  union { v16bf v; bf16x8 h[2]; } u;
  u.h[0] = *(const bf16x8*)(p);
  u.h[1] = *(const bf16x8*)(p + 8);
  return u.v;
}
// B operand (32x16 bf16), same contiguous k order as A: lane col = lane&15; lanes 0-15 hold K=0..15,
// lanes 16-31 hold K=16..31 -> 16 contiguous elements at p.
__device__ __forceinline__ v16bf ld_frag_b(const __bf16* p) {
  union { v16bf v; bf16x8 h[2]; } u;
  u.h[0] = *(const bf16x8*)(p);
  u.h[1] = *(const bf16x8*)(p + 8);
  return u.v;
}

__device__ __forceinline__ v8f wmma_bf16(v16bf a, v16bf b, v8f c) {
  return __builtin_amdgcn_wmma_f32_16x16x32_bf16(false, a, false, b, (short)0, c,
                                                 false, false);
}

// ---- Kernel 1: twiddle tables (bf16 hi/lo split) ---------------------------
// A1 [32][2048]: rows 0..15 = cos(2*pi*n*mi/2048), rows 16..31 = -sin(...)
// A3 [2048][32]: A3[n][j] = A1[j][n]   (same values, output-side layout)
// F  [64][16]  : head-dim twiddles cos/sin(2*pi*d*mj/64)
__global__ void spm_tables(__bf16* __restrict__ A1hi, __bf16* __restrict__ A1lo,
                           __bf16* __restrict__ A3hi, __bf16* __restrict__ A3lo,
                           float* __restrict__ Fc, float* __restrict__ Fs) {
  const int gid = blockIdx.x * blockDim.x + threadIdx.x;   // 0..65535
  const int j = gid >> 11;          // 0..31
  const int n = gid & 2047;
  const int mi = j & 15;
  const int pr = (n * mi) & 2047;   // exact angle reduction
  const float th = (float)pr * (TWO_PI / 2048.0f);
  const float val = (j < 16) ? cosf(th) : -sinf(th);
  const __bf16 hi = (__bf16)val;
  const __bf16 lo = (__bf16)(val - (float)hi);
  A1hi[j * 2048 + n] = hi;  A1lo[j * 2048 + n] = lo;
  A3hi[n * 32 + j]   = hi;  A3lo[n * 32 + j]   = lo;
  if (gid < 1024) {
    const int d = gid >> 4, mj = gid & 15;
    const int p2 = (d * mj) & 63;
    const float ph = (float)p2 * (TWO_PI / 64.0f);
    Fc[gid] = cosf(ph);
    Fs[gid] = sinf(ph);
  }
}

// ---- Kernel 2: effective complex filter per (b,h) --------------------------
// W_eff[b,h][mi][mj] = c_mj/(N*DH) * (Wpos[h,mi,mj] + mix[b,h]*Wneg[h,mi,mj])
// mix[b,h] = softmax(mixer[b,:])[h]  (the right-aligned 5-D broadcast).
__global__ void spm_weff(const float* __restrict__ pos, const float* __restrict__ neg,
                         const float* __restrict__ mixer, float* __restrict__ Weff) {
  const int gid = blockIdx.x * blockDim.x + threadIdx.x;   // 0..65535
  const int b  = gid >> 12;
  const int h  = (gid >> 8) & 15;
  const int mi = (gid >> 4) & 15;
  const int mj = gid & 15;
  float mx = mixer[b * 16];
  for (int t = 1; t < 16; ++t) mx = fmaxf(mx, mixer[b * 16 + t]);
  float s = 0.f;
  for (int t = 0; t < 16; ++t) s += __expf(mixer[b * 16 + t] - mx);
  const float mix = __expf(mixer[b * 16 + h] - mx) / s;
  const size_t pidx = (((size_t)h * 16 + mi) * 64 + mj) * 2;  // (H,M,DH,2)
  const float scale = (mj == 0 ? 1.0f : 2.0f) * (1.0f / (2048.0f * 64.0f));
  Weff[(size_t)gid * 2]     = scale * (pos[pidx]     + mix * neg[pidx]);
  Weff[(size_t)gid * 2 + 1] = scale * (pos[pidx + 1] + mix * neg[pidx + 1]);
}

// ---- Kernel 3: main. One workgroup (8 wave32) per (tensor,b,h). ------------
__global__ __launch_bounds__(256) void spm_main(
    const float* __restrict__ qg, const float* __restrict__ kg,
    const __bf16* __restrict__ A1hi, const __bf16* __restrict__ A1lo,
    const __bf16* __restrict__ A3hi, const __bf16* __restrict__ A3lo,
    const float* __restrict__ FcG, const float* __restrict__ FsG,
    const float* __restrict__ Weff, float* __restrict__ out) {
  // LDS: q chunk transposed [d=64][k=32] bf16 hi/lo, stride 48 (96B, 16B-aligned,
  // 2-way-max bank spread).  Reused as transposed Bp [d=64][j=32] for stage C.
  __shared__ __bf16 ldsHi[64 * 48];
  __shared__ __bf16 ldsLo[64 * 48];
  __shared__ float  Gbuf[32 * 64];
  __shared__ float  Yre[256], Yim[256];
  __shared__ float  Fc[1024], Fs[1024];

  const int tid    = threadIdx.x;
  const int lane   = tid & 31;
  const int wave   = tid >> 5;
  const int col    = lane & 15;
  const int half   = lane >> 4;        // 0/1
  const int abase  = half * 16;        // A-fragment K base
  const int bbase  = half * 16;        // B-fragment K base
  const int bh     = blockIdx.x & 255;
  const int tensor = blockIdx.x >> 8;

  const float* src = (tensor ? kg : qg) + (size_t)bh * (SEQ_N * DH_N);
  float* dst = out + (size_t)tensor * (256ull * SEQ_N * DH_N)
                   + (size_t)bh * (SEQ_N * DH_N);

  for (int i = tid; i < 1024; i += 256) { Fc[i] = FcG[i]; Fs[i] = FsG[i]; }

  // ======== Stage A: G(32x64) = A1(32x2048) * q(2048x64), bf16 WMMA =========
  const int tm = wave >> 2;            // 0..1  (G row tile)
  const int td = wave & 3;             // 0..3  (G col tile)
  const __bf16* a1h_row = A1hi + (tm * 16 + col) * 2048;
  const __bf16* a1l_row = A1lo + (tm * 16 + col) * 2048;

  v8f acc = {};
  for (int ks = 0; ks < 64; ++ks) {
    const int k0 = ks * 32;
    __syncthreads();                   // prior compute done before overwrite
    // cooperative fill: 2048 floats, fully coalesced float4, transposed +
    // split into bf16 hi/lo
#pragma unroll
    for (int hphase = 0; hphase < 2; ++hphase) {
      const int idx = hphase * 1024 + tid * 4;
      const int kr = idx >> 6;         // 0..31 (k within chunk)
      const int d  = idx & 63;
      const float4 v = *(const float4*)(src + (size_t)(k0 + kr) * 64 + d);
      const float vv[4] = {v.x, v.y, v.z, v.w};
#pragma unroll
      for (int i2 = 0; i2 < 4; ++i2) {
        const __bf16 h = (__bf16)vv[i2];
        ldsHi[(d + i2) * 48 + kr] = h;
        ldsLo[(d + i2) * 48 + kr] = (__bf16)(vv[i2] - (float)h);
      }
    }
    if (ks < 63)  // warm L2/L0 for the next chunk (global_prefetch_b8)
      __builtin_prefetch(src + (size_t)(k0 + 32 + (tid >> 3)) * 64 + (tid & 7) * 8, 0, 1);
    __syncthreads();

    const v16bf ah = ld_frag_a(a1h_row + k0 + abase);
    const v16bf al = ld_frag_a(a1l_row + k0 + abase);
    const v16bf bhv = ld_frag_b(ldsHi + (td * 16 + col) * 48 + bbase);
    const v16bf blv = ld_frag_b(ldsLo + (td * 16 + col) * 48 + bbase);
    acc = wmma_bf16(ah, bhv, acc);     // hi*hi
    acc = wmma_bf16(ah, blv, acc);     // hi*lo
    acc = wmma_bf16(al, bhv, acc);     // lo*hi  (~fp32-accurate split product)
  }
  {
    const int gr = tm * 16 + half * 8;
    const int gc = td * 16 + col;
#pragma unroll
    for (int v = 0; v < 8; ++v) Gbuf[(gr + v) * 64 + gc] = acc[v];
  }
  __syncthreads();

  // ======== Stage B: tiny mode-space step (VALU) ============================
  // B1: Qhat = G * conj(F); Y = Qhat o W_eff[b,h]
  {
    const int mi = tid >> 4, mj = tid & 15;
    float qre = 0.f, qim = 0.f;
    for (int d = 0; d < 64; ++d) {
      const float gr = Gbuf[mi * 64 + d], gi = Gbuf[(16 + mi) * 64 + d];
      const float c = Fc[d * 16 + mj],   s = Fs[d * 16 + mj];
      qre += gr * c + gi * s;
      qim += gi * c - gr * s;
    }
    const float wre = Weff[((size_t)bh * 256 + tid) * 2];
    const float wim = Weff[((size_t)bh * 256 + tid) * 2 + 1];
    Yre[tid] = qre * wre - qim * wim;
    Yim[tid] = qre * wim + qim * wre;
  }
  __syncthreads();
  // B2: P(16x64) = Y * F^T; store transposed Bp rows [Pre; Pim] as bf16 hi/lo
#pragma unroll
  for (int r = 0; r < 4; ++r) {
    const int p = tid + r * 256;
    const int mi = p >> 6, d = p & 63;
    float pre = 0.f, pim = 0.f;
    for (int mj = 0; mj < 16; ++mj) {
      const float yr = Yre[mi * 16 + mj], yi = Yim[mi * 16 + mj];
      const float c = Fc[d * 16 + mj],   s = Fs[d * 16 + mj];
      pre += yr * c - yi * s;
      pim += yr * s + yi * c;
    }
    const __bf16 h1 = (__bf16)pre;
    ldsHi[d * 48 + mi]      = h1;
    ldsLo[d * 48 + mi]      = (__bf16)(pre - (float)h1);
    const __bf16 h2 = (__bf16)pim;
    ldsHi[d * 48 + 16 + mi] = h2;
    ldsLo[d * 48 + 16 + mi] = (__bf16)(pim - (float)h2);
  }
  __syncthreads();

  // ======== Stage C: out(2048x64) = A3(2048x32) * Bp(32x64), bf16 WMMA ======
  for (int dt = 0; dt < 4; ++dt) {
    const v16bf bhv = ld_frag_b(ldsHi + (dt * 16 + col) * 48 + bbase);
    const v16bf blv = ld_frag_b(ldsLo + (dt * 16 + col) * 48 + bbase);
    for (int mt = 0; mt < 16; ++mt) {
      const int nt = wave * 16 + mt;   // 0..127
      const int n  = nt * 16 + col;
      const v16bf ah = ld_frag_a(A3hi + n * 32 + abase);
      const v16bf al = ld_frag_a(A3lo + n * 32 + abase);
      v8f c = {};
      c = wmma_bf16(ah, bhv, c);
      c = wmma_bf16(ah, blv, c);
      c = wmma_bf16(al, bhv, c);
      float* op = dst + (size_t)(nt * 16 + half * 8) * 64 + dt * 16 + col;
#pragma unroll
      for (int v = 0; v < 8; ++v) op[(size_t)v * 64] = c[v];
    }
  }
}

extern "C" void kernel_launch(void* const* d_in, const int* in_sizes, int n_in,
                              void* d_out, int out_size, void* d_ws, size_t ws_size,
                              hipStream_t stream) {
  (void)in_sizes; (void)n_in; (void)out_size; (void)ws_size;
  const float* q     = (const float*)d_in[0];
  const float* k     = (const float*)d_in[1];
  const float* pos   = (const float*)d_in[2];
  const float* neg   = (const float*)d_in[3];
  const float* mixer = (const float*)d_in[4];
  float* out = (float*)d_out;

  // Workspace layout (~1.01 MB, rebuilt every call -> deterministic)
  char* w = (char*)d_ws;
  __bf16* A1hi = (__bf16*)(w + 0);        // 32*2048 bf16 = 128KB
  __bf16* A1lo = (__bf16*)(w + 131072);
  __bf16* A3hi = (__bf16*)(w + 262144);   // 2048*32
  __bf16* A3lo = (__bf16*)(w + 393216);
  float*  Fc   = (float*)(w + 524288);    // 64*16 f32
  float*  Fs   = (float*)(w + 528384);
  float*  Weff = (float*)(w + 532480);    // 256*256*2 f32 = 512KB

  spm_tables<<<256, 256, 0, stream>>>(A1hi, A1lo, A3hi, A3lo, Fc, Fs);
  spm_weff<<<256, 256, 0, stream>>>(pos, neg, mixer, Weff);
  spm_main<<<512, 256, 0, stream>>>(q, k, A1hi, A1lo, A3hi, A3lo, Fc, Fs, Weff, out);
}
